// CRMF_50087908606248
// MI455X (gfx1250) — compile-verified, matrix operand order checked by reading, not run on hardware
//
#include <hip/hip_runtime.h>
#include <hip/hip_bf16.h>

typedef _Float16 h16;
typedef __attribute__((ext_vector_type(16))) _Float16 v16h;
typedef __attribute__((ext_vector_type(8)))  float    v8f;
typedef __attribute__((ext_vector_type(4)))  unsigned v4u;
typedef __attribute__((ext_vector_type(8)))  unsigned v8u;

#define WMMA_F16(a, b, c) \
  __builtin_amdgcn_wmma_f32_16x16x32_f16(false, (a), false, (b), (short)0, (c), false, false)

__device__ __forceinline__ float lrelu(float v) { return v > 0.f ? v : 0.2f * v; }

// ---------------------------------------------------------------------------
// Tensor Data Mover helpers (CDNA5 ISA).
// D# group0: count=1 | lds_addr | global_addr[56:0] | type=2
// D# group1: data_size/pad cfg | tensor_dim0/1 | tile_dim0/1 | dim0 stride
// ---------------------------------------------------------------------------
__device__ __forceinline__ v4u tdm_g0(unsigned lds_off, unsigned long long gaddr) {
  v4u g;
  g[0] = 1u;                                                   // count=1, user desc
  g[1] = lds_off;                                              // LDS byte address
  g[2] = (unsigned)gaddr;                                      // global addr lo
  g[3] = (unsigned)((gaddr >> 32) & 0x1FFFFFFull) | (2u << 30);// addr[56:32] | type=2
  return g;
}

__device__ __forceinline__ v8u tdm_g1(unsigned cfg, unsigned dim0, unsigned dim1,
                                      unsigned tile0, unsigned tile1, unsigned stride0) {
  v8u g;
  g[0] = cfg;                                          // wg_mask=0 | data_size | pad cfg
  g[1] = (dim0 & 0xFFFFu) << 16;                       // atomic_barrier=0 | dim0 lo16
  g[2] = (dim0 >> 16) | ((dim1 & 0xFFFFu) << 16);      // dim0 hi16 | dim1 lo16
  g[3] = (dim1 >> 16) | (tile0 << 16);                 // dim1 hi16 | tile_dim0
  g[4] = tile1 & 0xFFFFu;                              // tile_dim1 | tile_dim2=0
  g[5] = stride0;                                      // tensor_dim0_stride lo32
  g[6] = 0u;                                           // stride0 hi16 | stride1 lo16
  g[7] = 0u;
  return g;
}

__device__ __forceinline__ void tdm_load(v4u g0, v8u g1) {
  asm volatile("tensor_load_to_lds %0, %1" :: "s"(g0), "s"(g1) : "memory");
}
__device__ __forceinline__ void tdm_store(v4u g0, v8u g1) {
  asm volatile("tensor_store_from_lds %0, %1" :: "s"(g0), "s"(g1) : "memory");
}
__device__ __forceinline__ void tdm_wait() { __builtin_amdgcn_s_wait_tensorcnt(0); }

// data_size codes: 0=1B 1=2B 2=4B 3=8B
#define TDM_CFG_F32        (2u << 16)
#define TDM_CFG_F16        (1u << 16)
// pad_enable | pad_interval=6 (128 DWORDs = one 256-half row) | pad_amount=0 (1 DWORD)
#define TDM_CFG_F16_PAD258 ((1u << 16) | (1u << 20) | (6u << 22))

// ---------------------------------------------------------------------------
// Layer 0: fused instancenorm + projection + additive-attention + softmax +
// P@H + bias + transpose-concat + ELU.  One block per (batch, head).
// TDM loads x tile in; TDM stores the f16 activation slice out.
// ---------------------------------------------------------------------------
__global__ __launch_bounds__(256) void gat0_kernel(
    const float* __restrict__ x, const float* __restrict__ w0,
    const float* __restrict__ a_src0, const float* __restrict__ a_dst0,
    const float* __restrict__ b0, h16* __restrict__ hmid)
{
  extern __shared__ char smem[];
  const int b = blockIdx.x, head = blockIdx.y;
  const int t = threadIdx.x;
  const int wave = t >> 5, lane = t & 31, lm = lane & 15, lh = lane >> 4;

  float* xs    = (float*)(smem + 0);        // [256][32] f32 staging (TDM dest)
  h16*   outb  = (h16*)  (smem + 0);        // reuse: [256][64] f16 out staging
  float* meanv = (float*)(smem + 32768);    // [32]
  float* invv  = (float*)(smem + 32896);    // [32]
  h16*   xn    = (h16*)  (smem + 33024);    // [256][34] normalized f16
  h16*   w0t   = (h16*)  (smem + 50432);    // [64][32]  W0^T (n-major)
  h16*   ht    = (h16*)  (smem + 54528);    // [64][264] H^T (o-major)
  float* avs   = (float*)(smem + 88320);    // [64]
  float* avd   = (float*)(smem + 88576);    // [64]
  float* bias  = (float*)(smem + 88832);    // [64]
  float* ssrc  = (float*)(smem + 89088);    // [256]
  float* sdst  = (float*)(smem + 90112);    // [256]
  float* rmax  = (float*)(smem + 91136);    // [256]
  float* rden  = (float*)(smem + 92160);    // [256]
  float* smax  = (float*)(smem + 93184);    // [1]

  const int XN_S = 34, HT_S = 264;

  // ---- TDM: x[b] (256x32 f32 tile) -> LDS; other waves stage weights --
  if (wave == 0) {
    const unsigned long long xg = (unsigned long long)(x + (size_t)b * (256 * 32));
    tdm_load(tdm_g0(0u, xg), tdm_g1(TDM_CFG_F32, 32u, 256u, 32u, 256u, 32u));
  }
  const float* wg = w0 + (size_t)head * (32 * 64);
  for (int i = t; i < 32 * 64; i += 256) {
    int k = i >> 6, n = i & 63;
    w0t[n * 32 + k] = (h16)wg[i];        // transpose so B-frag K-pairs contiguous
  }
  if (t < 64) { avs[t] = a_src0[head * 64 + t]; avd[t] = a_dst0[head * 64 + t]; bias[t] = b0[t]; }
  if (wave == 0) tdm_wait();
  __syncthreads();

  // ---- instance norm over n per channel (32 channels) -----------------
  if (t < 32) {
    float s = 0.f, ss = 0.f;
    for (int r = 0; r < 256; ++r) { float v = xs[r * 32 + t]; s += v; ss += v * v; }
    float mu  = s * (1.f / 256.f);
    float var = ss * (1.f / 256.f) - mu * mu;
    meanv[t] = mu; invv[t] = rsqrtf(var + 1e-5f);
  }
  __syncthreads();
  for (int i = t; i < 256 * 32; i += 256) {
    int r = i >> 5, c = i & 31;
    xn[r * XN_S + c] = (h16)((xs[i] - meanv[c]) * invv[c]);
  }
  __syncthreads();

  // ---- projection H = xn(256x32) @ W0(32x64), store transposed --------
  for (int mi = 0; mi < 2; ++mi) {
    const int mt = wave * 2 + mi;
    const int row = mt * 16 + lm;
    v16h a;
#pragma unroll
    for (int v = 0; v < 8; ++v) {
      const int k0 = ((v & 4) ? 16 : 0) + lh * 8 + (v & 3) * 2;
      a[2 * v]     = xn[row * XN_S + k0];
      a[2 * v + 1] = xn[row * XN_S + k0 + 1];
    }
#pragma unroll
    for (int nt = 0; nt < 4; ++nt) {
      const int n = nt * 16 + lm;
      v16h bb;
#pragma unroll
      for (int v = 0; v < 8; ++v) {
        const int k0 = ((v & 4) ? 16 : 0) + lh * 8 + (v & 3) * 2;
        bb[2 * v]     = w0t[n * 32 + k0];
        bb[2 * v + 1] = w0t[n * 32 + k0 + 1];
      }
      v8f c = {};
      c = WMMA_F16(a, bb, c);
#pragma unroll
      for (int r = 0; r < 8; ++r)
        ht[n * HT_S + mt * 16 + r + 8 * lh] = (h16)c[r];
    }
  }
  __syncthreads();

  // ---- attention scalars: s_src, s_dst, rowmax, denom -----------------
  {
    float s0 = 0.f, s1 = 0.f;
    for (int o = 0; o < 64; ++o) {
      float hv = (float)ht[o * HT_S + t];
      s0 += hv * avs[o]; s1 += hv * avd[o];
    }
    ssrc[t] = s0; sdst[t] = s1;
  }
  __syncthreads();
  if (t == 0) {
    float m = sdst[0];
    for (int j = 1; j < 256; ++j) m = fmaxf(m, sdst[j]);
    smax[0] = m;                 // lrelu monotone => rowmax[i] = lrelu(s_src[i]+m)
  }
  __syncthreads();
  {
    const float rm = lrelu(ssrc[t] + smax[0]);
    float d = 0.f;
    for (int j = 0; j < 256; ++j) d += __expf(lrelu(ssrc[t] + sdst[j]) - rm);
    rmax[t] = rm; rden[t] = 1.f / d;
  }
  __syncthreads();   // also: all xs reads long done -> outb reuse is safe

  // ---- out = softmax(P) @ H ; fused bias + ELU -> LDS staging ---------
  for (int mi = 0; mi < 2; ++mi) {
    const int mt = wave * 2 + mi;
    const int row = mt * 16 + lm;
    const float sr = ssrc[row], rm = rmax[row], dn = rden[row];
    v8f acc[4] = {};
    for (int ks = 0; ks < 8; ++ks) {
      v16h a;  // P tile generated in registers per documented A-matrix layout
#pragma unroll
      for (int v = 0; v < 8; ++v) {
        const int k0 = ks * 32 + ((v & 4) ? 16 : 0) + lh * 8 + (v & 3) * 2;
        a[2 * v]     = (h16)(__expf(lrelu(sr + sdst[k0])     - rm) * dn);
        a[2 * v + 1] = (h16)(__expf(lrelu(sr + sdst[k0 + 1]) - rm) * dn);
      }
#pragma unroll
      for (int nt = 0; nt < 4; ++nt) {
        const int n = nt * 16 + lm;
        v16h bb;
#pragma unroll
        for (int v = 0; v < 8; ++v) {
          const int k0 = ks * 32 + ((v & 4) ? 16 : 0) + lh * 8 + (v & 3) * 2;
          bb[2 * v]     = ht[n * HT_S + k0];
          bb[2 * v + 1] = ht[n * HT_S + k0 + 1];
        }
        acc[nt] = WMMA_F16(a, bb, acc[nt]);
      }
    }
#pragma unroll
    for (int nt = 0; nt < 4; ++nt) {
      const int col = nt * 16 + lm;
      const float bv = bias[col];
#pragma unroll
      for (int r = 0; r < 8; ++r) {
        const int rr = mt * 16 + r + 8 * lh;
        float v = acc[nt][r] + bv;
        v = v > 0.f ? v : (__expf(v) - 1.f);   // ELU
        outb[rr * 64 + col] = (h16)v;
      }
    }
  }
  __syncthreads();

  // ---- TDM: LDS [256][64] f16 -> hmid[b][:, head*64 : head*64+64] -----
  if (wave == 0) {
    const unsigned long long og =
        (unsigned long long)(hmid + ((size_t)b * 256) * 256 + head * 64);
    tdm_store(tdm_g0(0u, og), tdm_g1(TDM_CFG_F16, 256u, 256u, 64u, 256u, 256u));
    tdm_wait();
  }
}

// ---------------------------------------------------------------------------
// Layer 1: fused instancenorm + projection + attention + bias.
// One block per batch.  TDM loads the 256x256 f16 tile (with 1-DWORD row
// padding -> 258-half LDS stride), TDM stores the fp32 result tile.
// ---------------------------------------------------------------------------
__global__ __launch_bounds__(256) void gat1_kernel(
    const h16* __restrict__ hmid, const float* __restrict__ w1,
    const float* __restrict__ a_src1, const float* __restrict__ a_dst1,
    const float* __restrict__ b1, float* __restrict__ out)
{
  extern __shared__ char smem[];
  const int b = blockIdx.x;
  const int t = threadIdx.x;
  const int wave = t >> 5, lane = t & 31, lm = lane & 15, lh = lane >> 4;

  h16*   hn    = (h16*)  (smem + 0);        // [256][258] staged + normalized
  float* outb  = (float*)(smem + 0);        // reuse: [256][32] f32 out staging
  float* meanv = (float*)(smem + 132096);   // [256]
  float* invv  = (float*)(smem + 133120);   // [256]
  h16*   w1t   = (h16*)  (smem + 134144);   // [32][258] W1^T
  h16*   h1t   = (h16*)  (smem + 150656);   // [32][264] H1^T
  float* a1s   = (float*)(smem + 167552);   // [32]
  float* a1d   = (float*)(smem + 167680);   // [32]
  float* bias  = (float*)(smem + 167808);   // [32]
  float* ssrc  = (float*)(smem + 167936);   // [256]
  float* sdst  = (float*)(smem + 168960);   // [256]
  float* rmax  = (float*)(smem + 169984);   // [256]
  float* rden  = (float*)(smem + 171008);   // [256]
  float* smax  = (float*)(smem + 172032);   // [1]

  const int HN_S = 258, W1_S = 258, H1T_S = 264;

  // ---- TDM: hmid[b] (256x256 f16) -> LDS with row padding -------------
  if (wave == 0) {
    const unsigned long long hg = (unsigned long long)(hmid + (size_t)b * 256 * 256);
    tdm_load(tdm_g0(0u, hg), tdm_g1(TDM_CFG_F16_PAD258, 256u, 256u, 256u, 256u, 256u));
  }
  for (int i = t; i < 256 * 32; i += 256) {
    int k = i >> 5, n = i & 31;
    w1t[n * W1_S + k] = (h16)w1[i];
  }
  if (t < 32) { a1s[t] = a_src1[t]; a1d[t] = a_dst1[t]; bias[t] = b1[t]; }
  if (wave == 0) tdm_wait();
  __syncthreads();

  // ---- instance norm over n per channel (256 channels, 1 per thread) --
  {
    float s = 0.f, ss = 0.f;
    for (int r = 0; r < 256; ++r) { float v = (float)hn[r * HN_S + t]; s += v; ss += v * v; }
    float mu  = s * (1.f / 256.f);
    float var = ss * (1.f / 256.f) - mu * mu;
    meanv[t] = mu; invv[t] = rsqrtf(var + 1e-5f);
  }
  __syncthreads();
  for (int i = t; i < 256 * 256; i += 256) {
    int r = i >> 8, c = i & 255;
    hn[r * HN_S + c] = (h16)(((float)hn[r * HN_S + c] - meanv[c]) * invv[c]);
  }
  __syncthreads();

  // ---- projection H1 = hn(256x256) @ W1(256x32), store transposed -----
  for (int mi = 0; mi < 2; ++mi) {
    const int mt = wave * 2 + mi;
    const int row = mt * 16 + lm;
    v8f acc[2] = {};
    for (int ks = 0; ks < 8; ++ks) {
      v16h a;
#pragma unroll
      for (int v = 0; v < 8; ++v) {
        const int k0 = ks * 32 + ((v & 4) ? 16 : 0) + lh * 8 + (v & 3) * 2;
        a[2 * v]     = hn[row * HN_S + k0];
        a[2 * v + 1] = hn[row * HN_S + k0 + 1];
      }
#pragma unroll
      for (int nt = 0; nt < 2; ++nt) {
        const int n = nt * 16 + lm;
        v16h bb;
#pragma unroll
        for (int v = 0; v < 8; ++v) {
          const int k0 = ks * 32 + ((v & 4) ? 16 : 0) + lh * 8 + (v & 3) * 2;
          bb[2 * v]     = w1t[n * W1_S + k0];
          bb[2 * v + 1] = w1t[n * W1_S + k0 + 1];
        }
        acc[nt] = WMMA_F16(a, bb, acc[nt]);
      }
    }
#pragma unroll
    for (int nt = 0; nt < 2; ++nt) {
      const int col = nt * 16 + lm;
#pragma unroll
      for (int r = 0; r < 8; ++r)
        h1t[col * H1T_S + mt * 16 + r + 8 * lh] = (h16)acc[nt][r];
    }
  }
  __syncthreads();

  // ---- attention scalars ----------------------------------------------
  {
    float s0 = 0.f, s1 = 0.f;
    for (int o = 0; o < 32; ++o) {
      float hv = (float)h1t[o * H1T_S + t];
      s0 += hv * a1s[o]; s1 += hv * a1d[o];
    }
    ssrc[t] = s0; sdst[t] = s1;
  }
  __syncthreads();
  if (t == 0) {
    float m = sdst[0];
    for (int j = 1; j < 256; ++j) m = fmaxf(m, sdst[j]);
    smax[0] = m;
  }
  __syncthreads();
  {
    const float rm = lrelu(ssrc[t] + smax[0]);
    float d = 0.f;
    for (int j = 0; j < 256; ++j) d += __expf(lrelu(ssrc[t] + sdst[j]) - rm);
    rmax[t] = rm; rden[t] = 1.f / d;
  }
  __syncthreads();   // hn reads done (projection) -> outb reuse safe

  // ---- out = softmax(P) @ H1 + bias -> LDS staging --------------------
  for (int mi = 0; mi < 2; ++mi) {
    const int mt = wave * 2 + mi;
    const int row = mt * 16 + lm;
    const float sr = ssrc[row], rm = rmax[row], dn = rden[row];
    v8f acc[2] = {};
    for (int ks = 0; ks < 8; ++ks) {
      v16h a;
#pragma unroll
      for (int v = 0; v < 8; ++v) {
        const int k0 = ks * 32 + ((v & 4) ? 16 : 0) + lh * 8 + (v & 3) * 2;
        a[2 * v]     = (h16)(__expf(lrelu(sr + sdst[k0])     - rm) * dn);
        a[2 * v + 1] = (h16)(__expf(lrelu(sr + sdst[k0 + 1]) - rm) * dn);
      }
#pragma unroll
      for (int nt = 0; nt < 2; ++nt) {
        const int n = nt * 16 + lm;
        v16h bb;
#pragma unroll
        for (int v = 0; v < 8; ++v) {
          const int k0 = ks * 32 + ((v & 4) ? 16 : 0) + lh * 8 + (v & 3) * 2;
          bb[2 * v]     = h1t[n * H1T_S + k0];
          bb[2 * v + 1] = h1t[n * H1T_S + k0 + 1];
        }
        acc[nt] = WMMA_F16(a, bb, acc[nt]);
      }
    }
#pragma unroll
    for (int nt = 0; nt < 2; ++nt) {
      const int col = nt * 16 + lm;
      const float bv = bias[col];
#pragma unroll
      for (int r = 0; r < 8; ++r) {
        const int rr = mt * 16 + r + 8 * lh;
        outb[rr * 32 + col] = acc[nt][r] + bv;
      }
    }
  }
  __syncthreads();

  // ---- TDM: LDS [256][32] f32 -> out[b] -------------------------------
  if (wave == 0) {
    const unsigned long long og = (unsigned long long)(out + (size_t)b * 256 * 32);
    tdm_store(tdm_g0(0u, og), tdm_g1(TDM_CFG_F32, 32u, 256u, 32u, 256u, 32u));
    tdm_wait();
  }
}

// ---------------------------------------------------------------------------
extern "C" void kernel_launch(void* const* d_in, const int* in_sizes, int n_in,
                              void* d_out, int out_size, void* d_ws, size_t ws_size,
                              hipStream_t stream) {
  (void)in_sizes; (void)n_in; (void)out_size; (void)ws_size;
  const float* x      = (const float*)d_in[0];
  const float* w0     = (const float*)d_in[1];
  const float* a_src0 = (const float*)d_in[2];
  const float* a_dst0 = (const float*)d_in[3];
  const float* b0     = (const float*)d_in[4];
  const float* w1     = (const float*)d_in[5];
  const float* a_src1 = (const float*)d_in[6];
  const float* a_dst1 = (const float*)d_in[7];
  const float* b1     = (const float*)d_in[8];
  float* out = (float*)d_out;
  h16*   hmid = (h16*)d_ws;   // [512][256][256] f16 = 64 MiB intermediate

  dim3 blk(256);
  dim3 g0(512, 4);
  gat0_kernel<<<g0, blk, 93248, stream>>>(x, w0, a_src0, a_dst0, b0, hmid);
  gat1_kernel<<<dim3(512), blk, 172096, stream>>>(hmid, w1, a_src1, a_dst1, b1, out);
}
